// LSTMCellLayerNorm_28656021799664
// MI455X (gfx1250) — matrix operand order repaired and checked by reading, not compiled, not run on hardware
//
#include <hip/hip_runtime.h>
#include <hip/hip_bf16.h>
#include <math.h>
#include <stdint.h>

// ---------------------------------------------------------------------------
// LayerNorm-LSTM cell for MI455X (gfx1250, wave32, WMMA, TDM)
//   Stage 1: split hx, W_hh into bf16 (hi, lo) pairs          [elementwise]
//   Stage 2: Y = hx @ W_hh^T via bf16x3 WMMA, fp32 accum
//            B tile staged in LDS by the Tensor Data Mover (double-buffered,
//            wave0-issued via a true scalar branch), falling back to
//            cooperative global->VGPR->LDS staging if TDM is unavailable.
//   Stage 3: gates = input + LN(Y); i,f,c,o; cy = LN(f*cx+i*c); hy = o*tanh(cy)
// ---------------------------------------------------------------------------

typedef __attribute__((ext_vector_type(16))) __bf16         v16bf;
typedef __attribute__((ext_vector_type(16))) unsigned short v16us;
typedef __attribute__((ext_vector_type(8)))  unsigned short v8us;
typedef __attribute__((ext_vector_type(8)))  float          v8f;
typedef __attribute__((ext_vector_type(4)))  float          v4f;
typedef __attribute__((ext_vector_type(4)))  unsigned int   v4u;
typedef __attribute__((ext_vector_type(8)))  int            v8i;
typedef __attribute__((ext_vector_type(4)))  int            v4i;

#if defined(__gfx1250__) && __has_builtin(__builtin_amdgcn_tensor_load_to_lds)
#define USE_TDM 1
#if __has_include(<hip/amd_detail/amd_gfx1250_TDM.h>)
#define TDM_6ARG 1   // therock-10.0 headers -> 6-arg builtin
#endif
#endif

#define LDS_STRIDE 40   // 32 data bf16 + 8 pad bf16 per row (bank-conflict-free)

// ---------------------------------------------------------------------------
// Stage 1: x -> (hi, lo) bf16 split.  x ~= hi + lo; the product expansion
// Ahi*Bhi + Ahi*Blo + Alo*Bhi carries ~16 mantissa bits per term (near-fp32).
// ---------------------------------------------------------------------------
__global__ __launch_bounds__(256) void split_convert(const float* __restrict__ x,
                                                     unsigned short* __restrict__ hi,
                                                     unsigned short* __restrict__ lo,
                                                     int n8) {
  const int i = blockIdx.x * 256 + threadIdx.x;
  if (i >= n8) return;
  const float* p = x + (size_t)i * 8;
  float a[8];
  *(v4f*)(a + 0) = *(const v4f*)(p + 0);
  *(v4f*)(a + 4) = *(const v4f*)(p + 4);
  v8us h, l;
#pragma unroll
  for (int e = 0; e < 8; ++e) {
    const unsigned ub = __builtin_bit_cast(unsigned, a[e]);
    const unsigned short hb = (unsigned short)(ub >> 16);      // truncate->bf16
    const float hf = __builtin_bit_cast(float, (unsigned)hb << 16);
    const float r = a[e] - hf;                                 // exact residual
    h[e] = hb;
    l[e] = (unsigned short)(__builtin_bit_cast(unsigned, r) >> 16);
  }
  *(v8us*)(hi + (size_t)i * 8) = h;
  *(v8us*)(lo + (size_t)i * 8) = l;
}

// ---------------------------------------------------------------------------
// TDM helpers (ISA ch.8: Tensor DMA Descriptor)
// ---------------------------------------------------------------------------
#if defined(USE_TDM)
static __device__ __forceinline__ unsigned lds_byte_offset(const void* p) {
  return (unsigned)(unsigned long long)(__attribute__((address_space(3))) const void*)p;
}

static __device__ __forceinline__ void tdm_wait0() {
#if __has_builtin(__builtin_amdgcn_s_wait_tensorcnt)
  __builtin_amdgcn_s_wait_tensorcnt(0);
#else
  asm volatile("s_wait_tensorcnt 0x0" ::: "memory");
#endif
}

// DMA a 64(row) x 32(K) bf16 tile, row stride strideD0 elements, into LDS at
// ldsOff with 4 DWORDs of padding after every 16 DWORDs (one 32-elem row)
// -> LDS row stride = LDS_STRIDE bf16.
static __device__ __forceinline__ void tdm_load_tile(const unsigned short* gsrc,
                                                     unsigned ldsOff,
                                                     int tensorD0, int tensorD1,
                                                     int strideD0) {
  const unsigned long long ga = (unsigned long long)(uintptr_t)gsrc;
  v4u g0;
  g0[0] = 1u;                                            // count=1, user D#
  g0[1] = ldsOff;                                        // lds_addr (bytes)
  g0[2] = (unsigned)ga;                                  // global_addr[31:0]
  g0[3] = (unsigned)((ga >> 32) & 0x01ffffffu)           // global_addr[56:32]
          | (2u << 30);                                  // type = 2 ("image")
  v8i g1;
  g1[0] = (int)((1u << 16)      // data_size = 2 bytes
                | (1u << 20)    // pad_enable
                | (3u << 22)    // pad_interval: 16 DWORDs (= one 64B row)
                | (3u << 25));  // pad_amount:   4 DWORDs (= 16B)
  g1[1] = (int)(((unsigned)tensorD0 & 0xffffu) << 16);                 // dim0[15:0]
  g1[2] = (int)((((unsigned)tensorD0 >> 16) & 0xffffu)                 // dim0[31:16]
                | (((unsigned)tensorD1 & 0xffffu) << 16));             // dim1[15:0]
  g1[3] = (int)((((unsigned)tensorD1 >> 16) & 0xffffu)                 // dim1[31:16]
                | (32u << 16));                                        // tile_dim0
  g1[4] = 64;                                                          // tile_dim1
  g1[5] = strideD0;                                                    // dim0_stride
  g1[6] = 0;
  g1[7] = 0;
  const v4i z4 = {0, 0, 0, 0};
#if defined(TDM_6ARG)
  const v8i z8 = {0, 0, 0, 0, 0, 0, 0, 0};
  __builtin_amdgcn_tensor_load_to_lds(g0, g1, z4, z4, z8, 0);
#else
  __builtin_amdgcn_tensor_load_to_lds(g0, g1, z4, z4, 0);
#endif
}
#endif  // USE_TDM

// ---------------------------------------------------------------------------
// Stage 2 GEMM: Y[M,N] = A[M,K] * W[N,K]^T  (M=8192, N=4096, K=1024)
// WG = 256 threads = 8 waves; WG tile 256x64, wave tile 32x64 (8x v8f accum).
// ---------------------------------------------------------------------------
__global__ __launch_bounds__(256) void ln_lstm_gemm_bf16x3(
    const unsigned short* __restrict__ Ahi, const unsigned short* __restrict__ Alo,
    const unsigned short* __restrict__ Whi, const unsigned short* __restrict__ Wlo,
    float* __restrict__ Y, int M, int N, int K) {
  const int tid  = threadIdx.x;
  const int lane = tid & 31;
  const int r    = lane & 15;   // A row / B column owned by this lane
  const int kh   = lane >> 4;   // lane-half -> K sub-range per ISA layout
  const int wave = tid >> 5;

  const int row0 = blockIdx.y * 256 + wave * 32;
  const int col0 = blockIdx.x * 64;

  __shared__ unsigned short lbh[2][64 * LDS_STRIDE];
  __shared__ unsigned short lbl[2][64 * LDS_STRIDE];

  // A 16x32 bf16 fragment layout: lane l holds row l&15,
  //   elems 0..7 = K (l>>4)*8 + 0..7 ; elems 8..15 = +16..23
  const unsigned short* a0h = Ahi + (size_t)(row0 + r) * K + kh * 8;
  const unsigned short* a0l = Alo + (size_t)(row0 + r) * K + kh * 8;
  const unsigned short* a1h = a0h + (size_t)16 * K;
  const unsigned short* a1l = a0l + (size_t)16 * K;

#if !defined(USE_TDM)
  // cooperative B staging: thread -> (W row, K chunk)
  const int wrow = tid >> 2;        // 0..63
  const int kc   = (tid & 3) * 8;   // 0,8,16,24
  const unsigned short* wgh = Whi + (size_t)(col0 + wrow) * K + kc;
  const unsigned short* wgl = Wlo + (size_t)(col0 + wrow) * K + kc;
#endif

  v8f acc[2][4];
#pragma unroll
  for (int g = 0; g < 2; ++g)
#pragma unroll
    for (int t = 0; t < 4; ++t) acc[g][t] = v8f{0.f};

#if defined(USE_TDM)
  // Wave id as an SGPR value -> compiler emits a true scalar branch, so only
  // wave 0 *issues* the DMA (TDM ignores EXEC, so exec-masking would not
  // prevent duplicate issues from the other 7 waves).
  const int wave_u = __builtin_amdgcn_readfirstlane(wave);
  if (wave_u == 0) {
    // Prologue: DMA first B tile (hi+lo) into buffer 0.
    tdm_load_tile(Whi + (size_t)col0 * K, lds_byte_offset(&lbh[0][0]), K, N, K);
    tdm_load_tile(Wlo + (size_t)col0 * K, lds_byte_offset(&lbl[0][0]), K, N, K);
  }
#endif

  for (int k0 = 0; k0 < K; k0 += 32) {
#if defined(USE_TDM)
    const int buf = (k0 >> 5) & 1;
    tdm_wait0();                     // scalar; no-op for waves with no TDM ops
    __syncthreads();                 // tile visible to all waves
    if (wave_u == 0 && (k0 + 32) < K) {
      // Issue next tile into the other buffer while everyone consumes `buf`.
      // Safe: a wave reaches this barrier only after its previous-iteration
      // ds_loads were consumed by WMMAs (dscnt waits precede them).
      const int nb = buf ^ 1;
      tdm_load_tile(Whi + (size_t)col0 * K + (k0 + 32),
                    lds_byte_offset(&lbh[nb][0]), K, N, K);
      tdm_load_tile(Wlo + (size_t)col0 * K + (k0 + 32),
                    lds_byte_offset(&lbl[nb][0]), K, N, K);
    }
#else
    const int buf = 0;
    const v8us gh = *(const v8us*)(wgh + k0);
    const v8us gl = *(const v8us*)(wgl + k0);
    __syncthreads();                 // previous iteration readers done
    *(v8us*)&lbh[0][wrow * LDS_STRIDE + kc] = gh;
    *(v8us*)&lbl[0][wrow * LDS_STRIDE + kc] = gl;
    __syncthreads();
#endif

    // ---- A fragments (global, L2-resident) ----
    v16us ah0, al0, ah1, al1;
    *((v8us*)&ah0 + 0) = *(const v8us*)(a0h + k0);
    *((v8us*)&ah0 + 1) = *(const v8us*)(a0h + k0 + 16);
    *((v8us*)&al0 + 0) = *(const v8us*)(a0l + k0);
    *((v8us*)&al0 + 1) = *(const v8us*)(a0l + k0 + 16);
    *((v8us*)&ah1 + 0) = *(const v8us*)(a1h + k0);
    *((v8us*)&ah1 + 1) = *(const v8us*)(a1h + k0 + 16);
    *((v8us*)&al1 + 0) = *(const v8us*)(a1l + k0);
    *((v8us*)&al1 + 1) = *(const v8us*)(a1l + k0 + 16);
    if (k0 + 32 < K) __builtin_prefetch((const void*)(a0h + k0 + 32), 0, 1);

    const v16bf A0h = __builtin_bit_cast(v16bf, ah0);
    const v16bf A0l = __builtin_bit_cast(v16bf, al0);
    const v16bf A1h = __builtin_bit_cast(v16bf, ah1);
    const v16bf A1l = __builtin_bit_cast(v16bf, al1);

    // B 32x16 bf16 fragment layout: lane l holds column l&15,
    //   elems 0..7 = K (l>>4)*8 + 0..7 ; elems 8..15 = +16..23 (2x ds_load_b128)
#pragma unroll
    for (int t = 0; t < 4; ++t) {
      const int rb = (t * 16 + r) * LDS_STRIDE + kh * 8;
      v16us bh, bl;
      *((v8us*)&bh + 0) = *(const v8us*)&lbh[buf][rb];
      *((v8us*)&bh + 1) = *(const v8us*)&lbh[buf][rb + 16];
      *((v8us*)&bl + 0) = *(const v8us*)&lbl[buf][rb];
      *((v8us*)&bl + 1) = *(const v8us*)&lbl[buf][rb + 16];
      const v16bf Bh = __builtin_bit_cast(v16bf, bh);
      const v16bf Bl = __builtin_bit_cast(v16bf, bl);

      acc[0][t] = __builtin_amdgcn_wmma_f32_16x16x32_bf16(
          false, A0h, false, Bh, (short)0, acc[0][t], false, false);
      acc[0][t] = __builtin_amdgcn_wmma_f32_16x16x32_bf16(
          false, A0h, false, Bl, (short)0, acc[0][t], false, false);
      acc[0][t] = __builtin_amdgcn_wmma_f32_16x16x32_bf16(
          false, A0l, false, Bh, (short)0, acc[0][t], false, false);
      acc[1][t] = __builtin_amdgcn_wmma_f32_16x16x32_bf16(
          false, A1h, false, Bh, (short)0, acc[1][t], false, false);
      acc[1][t] = __builtin_amdgcn_wmma_f32_16x16x32_bf16(
          false, A1h, false, Bl, (short)0, acc[1][t], false, false);
      acc[1][t] = __builtin_amdgcn_wmma_f32_16x16x32_bf16(
          false, A1l, false, Bh, (short)0, acc[1][t], false, false);
    }
  }

  // C/D layout: lane l -> N = col0 + t*16 + (l&15); VGPR v -> M = (l>>4)*8 + v
#pragma unroll
  for (int g = 0; g < 2; ++g)
#pragma unroll
    for (int t = 0; t < 4; ++t)
#pragma unroll
      for (int v = 0; v < 8; ++v) {
        const int m = row0 + g * 16 + kh * 8 + v;
        const int n = col0 + t * 16 + r;
        Y[(size_t)m * N + n] = acc[g][t][v];
      }
}

// ---------------------------------------------------------------------------
// Stage 3 epilogue: one 256-thread block per batch row.
// j = 4t + 1024*s puts the i/f/c/o gate values for hidden units 4t..4t+3
// into the thread's four float4 registers -> fully b128 loads/stores.
// ---------------------------------------------------------------------------
__global__ __launch_bounds__(256) void ln_lstm_epilogue(
    const float* __restrict__ Y,       // [B, 4H]
    const float* __restrict__ igates,  // [B, 4H]
    const float* __restrict__ cx,      // [B, H]
    const float* __restrict__ gamma_h, const float* __restrict__ beta_h,
    const float* __restrict__ gamma_c, const float* __restrict__ beta_c,
    float* __restrict__ hy, float* __restrict__ cy) {
  constexpr int H  = 1024;
  constexpr int H4 = 4096;
  const int b = blockIdx.x;
  const int t = threadIdx.x;

  __shared__ float s0[256];
  __shared__ float s1[256];

  const float* yrow = Y + (size_t)b * H4;
  v4f y4[4];
  float lsum = 0.f, lsq = 0.f;
#pragma unroll
  for (int s = 0; s < 4; ++s) {
    y4[s] = *(const v4f*)(yrow + 4 * t + H * s);
#pragma unroll
    for (int e = 0; e < 4; ++e) { lsum += y4[s][e]; lsq += y4[s][e] * y4[s][e]; }
  }
  s0[t] = lsum; s1[t] = lsq;
  __syncthreads();
#pragma unroll
  for (int off = 128; off > 0; off >>= 1) {
    if (t < off) { s0[t] += s0[t + off]; s1[t] += s1[t + off]; }
    __syncthreads();
  }
  const float mu  = s0[0] * (1.f / H4);
  const float var = s1[0] * (1.f / H4) - mu * mu;
  const float rs  = rsqrtf(var + 1e-5f);
  __syncthreads();  // all readers done before shared reuse

  const float* irow = igates + (size_t)b * H4;
  v4f g4[4];
#pragma unroll
  for (int s = 0; s < 4; ++s) {
    const int j0 = 4 * t + H * s;
    const v4f iv = *(const v4f*)(irow + j0);
    const v4f gm = *(const v4f*)(gamma_h + j0);
    const v4f bt = *(const v4f*)(beta_h + j0);
#pragma unroll
    for (int e = 0; e < 4; ++e)
      g4[s][e] = iv[e] + (y4[s][e] - mu) * rs * gm[e] + bt[e];
  }

  const v4f cx4 = *(const v4f*)(cx + (size_t)b * H + 4 * t);
  float cnew[4], og[4];
  float csum = 0.f, csq = 0.f;
#pragma unroll
  for (int e = 0; e < 4; ++e) {
    const float ig = 1.f / (1.f + __expf(-g4[0][e]));
    const float fg = 1.f / (1.f + __expf(-g4[1][e]));
    const float cg = tanhf(g4[2][e]);
    og[e] = 1.f / (1.f + __expf(-g4[3][e]));
    const float c = fg * cx4[e] + ig * cg;
    cnew[e] = c;
    csum += c;
    csq  += c * c;
  }
  s0[t] = csum; s1[t] = csq;
  __syncthreads();
#pragma unroll
  for (int off = 128; off > 0; off >>= 1) {
    if (t < off) { s0[t] += s0[t + off]; s1[t] += s1[t + off]; }
    __syncthreads();
  }
  const float muc  = s0[0] * (1.f / H);
  const float varc = s1[0] * (1.f / H) - muc * muc;
  const float rsc  = rsqrtf(varc + 1e-5f);

  const v4f gc4 = *(const v4f*)(gamma_c + 4 * t);
  const v4f bc4 = *(const v4f*)(beta_c + 4 * t);
  v4f cv4, hv4;
#pragma unroll
  for (int e = 0; e < 4; ++e) {
    const float cv = (cnew[e] - muc) * rsc * gc4[e] + bc4[e];
    cv4[e] = cv;
    hv4[e] = og[e] * tanhf(cv);
  }
  *(v4f*)(cy + (size_t)b * H + 4 * t) = cv4;
  *(v4f*)(hy + (size_t)b * H + 4 * t) = hv4;
}

// ---------------------------------------------------------------------------
// Inputs (setup_inputs order):
//   0 input [B,4H]  1 hx [B,H]  2 cx [B,H]  3 weight_hh [4H,H]
//   4 gamma_h [4H]  5 beta_h [4H]  6 gamma_c [H]  7 beta_c [H]
// d_out = hy [B,H] ++ cy [B,H].
// d_ws layout: Y fp32 [B,4H] | Ahi | Alo bf16 [B,H] | Whi | Wlo bf16 [4H,H]
//   = 128MB + 32MB + 16MB = 176MB for B=8192, H=1024.
// ---------------------------------------------------------------------------
extern "C" void kernel_launch(void* const* d_in, const int* in_sizes, int n_in,
                              void* d_out, int out_size, void* d_ws, size_t ws_size,
                              hipStream_t stream) {
  const float* input   = (const float*)d_in[0];
  const float* hx      = (const float*)d_in[1];
  const float* cx      = (const float*)d_in[2];
  const float* W       = (const float*)d_in[3];
  const float* gamma_h = (const float*)d_in[4];
  const float* beta_h  = (const float*)d_in[5];
  const float* gamma_c = (const float*)d_in[6];
  const float* beta_c  = (const float*)d_in[7];

  const int H  = in_sizes[6];       // 1024
  const int H4 = in_sizes[4];       // 4096
  const int B  = in_sizes[1] / H;   // 8192

  float* Y = (float*)d_ws;                                   // [B, 4H] fp32
  unsigned short* Ahi = (unsigned short*)(Y + (size_t)B * H4);
  unsigned short* Alo = Ahi + (size_t)B * H;
  unsigned short* Whi = Alo + (size_t)B * H;
  unsigned short* Wlo = Whi + (size_t)H4 * H;

  float* hy    = (float*)d_out;
  float* cyPtr = hy + (size_t)B * H;

  const int nA8 = B * H / 8;    // 8-element chunks
  const int nW8 = H4 * H / 8;
  split_convert<<<dim3((nA8 + 255) / 256), dim3(256), 0, stream>>>(hx, Ahi, Alo, nA8);
  split_convert<<<dim3((nW8 + 255) / 256), dim3(256), 0, stream>>>(W, Whi, Wlo, nW8);

  ln_lstm_gemm_bf16x3<<<dim3(H4 / 64, B / 256), dim3(256), 0, stream>>>(
      Ahi, Alo, Whi, Wlo, Y, B, H4, H);

  ln_lstm_epilogue<<<dim3(B), dim3(256), 0, stream>>>(
      Y, input, cx, gamma_h, beta_h, gamma_c, beta_c, hy, cyPtr);
}
